// MultiHeadSelfAttention_43413529427997
// MI455X (gfx1250) — matrix operand order repaired and checked by reading, not compiled, not run on hardware
//
#include <hip/hip_runtime.h>
#include <hip/hip_bf16.h>
#include <math.h>

// ---------------------------------------------------------------------------
// MultiHeadSelfAttention for MI455X (gfx1250, wave32, WMMA).
// Pipeline: prep (f32->bf16, weight transpose) -> QKV GEMMs (WMMA bf16)
//        -> flash attention (WMMA bf16, online softmax,
//           cooperative async K/V staging to LDS, LDS P-exchange)
//        -> output projection GEMM (WMMA bf16, fp32 out).
// ---------------------------------------------------------------------------

#define DEV __device__ __forceinline__

typedef __bf16 bf16_t;
typedef __bf16 v8bf  __attribute__((ext_vector_type(8)));
typedef __bf16 v16bf __attribute__((ext_vector_type(16)));
typedef float  v8f   __attribute__((ext_vector_type(8)));

// Types for the async-to-LDS builtin (signature revealed by the compiler:
// first param is `int __attribute__((vector_size(16)))` pointer in AS1).
typedef int v4i_vs __attribute__((vector_size(16)));
typedef __attribute__((address_space(1))) v4i_vs* g_v4i_ptr;
typedef __attribute__((address_space(3))) v4i_vs* l_v4i_ptr;

constexpr int Bn = 4;            // batch
constexpr int Tn = 2048;         // sequence
constexpr int Cn = 768;          // embed
constexpr int Hn = 12;           // heads
constexpr int Dn = 64;           // head dim
constexpr int Mn = Bn * Tn;      // 8192 token rows
constexpr int BH = Bn * Hn;      // 48

// ---- WMMA fragment loads (ISA 7.12.2 layouts, wave32) ----------------------
// A (16x32 bf16): row m = lane&15 (both halves). Lane holds:
//   elems 0..7  -> K = 8*half + 0..7 ; elems 8..15 -> K = 16 + 8*half + 0..7
DEV v16bf load_frag_a(const bf16_t* row_k0, int lane) {
  const int hf = (lane >> 4) & 1;
  v8bf lo = *reinterpret_cast<const v8bf*>(row_k0 + hf * 8);
  v8bf hi = *reinterpret_cast<const v8bf*>(row_k0 + 16 + hf * 8);
  return __builtin_shufflevector(lo, hi, 0,1,2,3,4,5,6,7,8,9,10,11,12,13,14,15);
}

// B (32x16 bf16): col n = lane&15. Lane holds elems 0..7 -> K = 8*half + 0..7,
// elems 8..15 -> K = 16 + 8*half + 0..7 (operand stored N-major with K contiguous).
DEV v16bf load_frag_b(const bf16_t* col_k0, int lane) {
  const int hf = (lane >> 4) & 1;
  const bf16_t* p = col_k0 + hf * 8;
  v8bf lo = *reinterpret_cast<const v8bf*>(p);
  v8bf hi = *reinterpret_cast<const v8bf*>(p + 16);
  return __builtin_shufflevector(lo, hi, 0,1,2,3,4,5,6,7,8,9,10,11,12,13,14,15);
}

DEV v8f wmma_bf16(v16bf a, v16bf b, v8f c) {
  return __builtin_amdgcn_wmma_f32_16x16x32_bf16(
      /*neg_a=*/false, a, /*neg_b=*/false, b,
      /*c_mod=*/(short)0, c, /*reuse_a=*/false, /*reuse_b=*/false);
}

// 16-byte global -> LDS stage; async DMA path on gfx1250 toolchains that
// expose the builtin, plain vector copy otherwise.
DEV void stage16(bf16_t* lds_dst, const bf16_t* gsrc) {
#if __has_builtin(__builtin_amdgcn_global_load_async_to_lds_b128)
  __builtin_amdgcn_global_load_async_to_lds_b128(
      (g_v4i_ptr)gsrc, (l_v4i_ptr)lds_dst, 0, 0);
#else
  *reinterpret_cast<v8bf*>(lds_dst) = *reinterpret_cast<const v8bf*>(gsrc);
#endif
}

DEV void stage_wait() {
#if __has_builtin(__builtin_amdgcn_s_wait_asynccnt)
  __builtin_amdgcn_s_wait_asynccnt(0);
#else
  asm volatile("s_wait_asynccnt 0" ::: "memory");
#endif
}

// ---- prep kernels ----------------------------------------------------------
__global__ void cvt_f32_bf16(const float* __restrict__ in,
                             bf16_t* __restrict__ out, int n) {
  int i = blockIdx.x * blockDim.x + threadIdx.x;
  if (i < n) out[i] = (bf16_t)in[i];
}

// w: [C][C] (k-major, x @ W) -> wt: [n][k] bf16 so B-fragments are contiguous.
__global__ void transpose_w_bf16(const float* __restrict__ w,
                                 bf16_t* __restrict__ wt) {
  int i = blockIdx.x * blockDim.x + threadIdx.x;
  if (i >= Cn * Cn) return;
  int n = i / Cn, k = i % Cn;
  wt[i] = (bf16_t)w[(size_t)k * Cn + n];
}

// ---- GEMM: out = A[M][C] x Wt[C][C]^T + bias -------------------------------
enum { MODE_QK = 0, MODE_VT = 1, MODE_F32 = 2 };

__global__ __launch_bounds__(256)
void gemm_bf16_wmma(const bf16_t* __restrict__ A,
                    const bf16_t* __restrict__ Wt,   // [n][k]
                    const float* __restrict__ bias,  // [C]
                    bf16_t* __restrict__ out_bf,
                    float* __restrict__ out_f,
                    int mode) {
  const int lane = threadIdx.x & 31;
  const int wave = threadIdx.x >> 5;
  const int tile = blockIdx.x * 8 + wave;
  const int tiles_n = Cn / 64;                    // 12
  const int m0 = (tile / tiles_n) * 64;
  const int n0 = (tile % tiles_n) * 64;
  const int hf = lane >> 4;
  const int nn = lane & 15;

  v8f acc[4][4];
#pragma unroll
  for (int i = 0; i < 4; ++i)
#pragma unroll
    for (int j = 0; j < 4; ++j) acc[i][j] = {};

  for (int k0 = 0; k0 < Cn; k0 += 32) {
    v16bf a[4], b[4];
#pragma unroll
    for (int mi = 0; mi < 4; ++mi)
      a[mi] = load_frag_a(A + (size_t)(m0 + mi * 16 + nn) * Cn + k0, lane);
#pragma unroll
    for (int ni = 0; ni < 4; ++ni)
      b[ni] = load_frag_b(Wt + (size_t)(n0 + ni * 16 + nn) * Cn + k0, lane);
#pragma unroll
    for (int mi = 0; mi < 4; ++mi)
#pragma unroll
      for (int ni = 0; ni < 4; ++ni)
        acc[mi][ni] = wmma_bf16(a[mi], b[ni], acc[mi][ni]);
  }

  // Epilogue. C-fragment: n = lane&15, row m = r + 8*half.
#pragma unroll
  for (int ni = 0; ni < 4; ++ni) {
    const int ng = n0 + ni * 16 + nn;
    const float bv = bias[ng];
#pragma unroll
    for (int mi = 0; mi < 4; ++mi) {
#pragma unroll
      for (int r = 0; r < 8; ++r) {
        const int mg = m0 + mi * 16 + r + hf * 8;   // token index
        const float v = acc[mi][ni][r] + bv;
        if (mode == MODE_F32) {
          out_f[(size_t)mg * Cn + ng] = v;
        } else {
          const int bb = mg >> 11, t = mg & (Tn - 1);
          const int h = ng >> 6, d = ng & (Dn - 1);
          size_t idx = (mode == MODE_QK)
                         ? (((size_t)(bb * Hn + h) * Tn + t) * Dn + d)   // [BH][T][D]
                         : (((size_t)(bb * Hn + h) * Dn + d) * Tn + t);  // [BH][D][T]
          out_bf[idx] = (bf16_t)v;
        }
      }
    }
  }
}

// ---- Flash attention -------------------------------------------------------
// One workgroup = 8 waves = 128 consecutive query rows of one (b,h).
// Each 32-key tile of K and V is staged into LDS ONCE (async DMA) and shared
// by all 8 waves; trailing waves skip masked tiles but join the barriers.
__global__ __launch_bounds__(256)
void flash_attn_wmma(const bf16_t* __restrict__ Q,   // [BH][T][D]
                     const bf16_t* __restrict__ K,   // [BH][T][D]
                     const bf16_t* __restrict__ Vt,  // [BH][D][T]
                     bf16_t* __restrict__ O) {       // [B][T][C]
  __shared__ bf16_t kbuf[32 * 64];                   // K tile  [key][d]   4 KB
  __shared__ bf16_t vbuf[64 * 32];                   // V tile  [d][key]   4 KB
  __shared__ bf16_t pbuf[8][16 * 32];                // per-wave P tiles   8 KB

  const int tid  = threadIdx.x;
  const int lane = tid & 31;
  const int wave = tid >> 5;
  const int bh   = blockIdx.x / (Tn / 128);
  const int qblk = blockIdx.x % (Tn / 128);
  const int q0b  = qblk * 128;                       // block's first q row
  const int q0   = q0b + wave * 16;                  // wave's first q row
  const int bb   = bh / Hn, h = bh % Hn;
  const int hf   = lane >> 4;
  const int nn   = lane & 15;

  // Q A-fragments, reused across the whole key loop (D=64 -> two K=32 chunks).
  const bf16_t* qrow = Q + ((size_t)bh * Tn + q0 + nn) * Dn;
  const v16bf qa0 = load_frag_a(qrow, lane);
  const v16bf qa1 = load_frag_a(qrow + 32, lane);

  v8f o[4];
#pragma unroll
  for (int f = 0; f < 4; ++f) o[f] = {};
  float rm[8], l[8];
#pragma unroll
  for (int r = 0; r < 8; ++r) { rm[r] = -INFINITY; l[r] = 0.0f; }

  const float scale = 0.125f;                        // 1/sqrt(64)
  const int nkt = (q0b + 128) / 32;                  // block causal tile count
  bf16_t* pl = pbuf[wave];

  // Per-thread staging coordinates (16 B per thread per tile).
  const int krow = tid >> 3, kch = tid & 7;          // K: 32 rows x 8 chunks
  const int vrow = tid >> 2, vch = tid & 3;          // V: 64 rows x 4 chunks

  for (int kt = 0; kt < nkt; ++kt) {
    const int kb0 = kt * 32;

    // ---- cooperative staging of K/V tiles into LDS (ASYNCcnt path) ----
    stage16(kbuf + krow * 64 + kch * 8,
            K + ((size_t)bh * Tn + kb0 + krow) * Dn + kch * 8);
    stage16(vbuf + vrow * 32 + vch * 8,
            Vt + ((size_t)bh * Dn + vrow) * Tn + kb0 + vch * 8);
    stage_wait();                                    // own async loads done
    __syncthreads();                                 // everyone's loads done

    if (kb0 <= q0 + 15) {                            // wave has live keys here
      // S = Q K^T for two 16-wide key sub-tiles (4 WMMAs), K from LDS.
      v8f s0 = {}, s1 = {};
      {
        const bf16_t* kr0 = kbuf + (size_t)nn * 64;
        const bf16_t* kr1 = kbuf + (size_t)(16 + nn) * 64;
        s0 = wmma_bf16(qa0, load_frag_b(kr0, lane), s0);
        s0 = wmma_bf16(qa1, load_frag_b(kr0 + 32, lane), s0);
        s1 = wmma_bf16(qa0, load_frag_b(kr1, lane), s1);
        s1 = wmma_bf16(qa1, load_frag_b(kr1 + 32, lane), s1);
      }

      // Causal mask + online softmax (row reductions across 16-lane half).
      float p0v[8], p1v[8];
#pragma unroll
      for (int r = 0; r < 8; ++r) {
        const int qg = q0 + r + hf * 8;
        float v0 = s0[r] * scale; if (kb0 + nn > qg)      v0 = -INFINITY;
        float v1 = s1[r] * scale; if (kb0 + 16 + nn > qg) v1 = -INFINITY;
        float mx = fmaxf(v0, v1);
#pragma unroll
        for (int d = 8; d >= 1; d >>= 1) mx = fmaxf(mx, __shfl_xor(mx, d, 32));
        const float mnew = fmaxf(rm[r], mx);
        const float corr = __expf(rm[r] - mnew);
        const float p0 = __expf(v0 - mnew);
        const float p1 = __expf(v1 - mnew);
        float rs = p0 + p1;
#pragma unroll
        for (int d = 8; d >= 1; d >>= 1) rs += __shfl_xor(rs, d, 32);
        l[r]  = l[r] * corr + rs;
        rm[r] = mnew;
        p0v[r] = p0; p1v[r] = p1;
#pragma unroll
        for (int f = 0; f < 4; ++f) o[f][r] *= corr; // rescale O accumulators
      }

      // C-layout P -> A-layout via per-wave LDS tile [16 rows][32 keys].
#pragma unroll
      for (int r = 0; r < 8; ++r) {
        const int row = r + hf * 8;
        pl[row * 32 + nn]      = (bf16_t)p0v[r];
        pl[row * 32 + 16 + nn] = (bf16_t)p1v[r];
      }
      asm volatile("s_wait_dscnt 0" ::: "memory");   // same-wave LDS RAW
      const v16bf pa = load_frag_a(pl + nn * 32, lane);

      // O += P V (4 WMMAs over the 64-wide head dim), V from LDS.
#pragma unroll
      for (int f = 0; f < 4; ++f)
        o[f] = wmma_bf16(pa, load_frag_b(vbuf + (size_t)(f * 16 + nn) * 32, lane),
                         o[f]);
    }
    __syncthreads();                                 // tiles reusable next iter
  }

  // Normalize and store bf16 [B][T][C], col = h*64 + f*16 + n.
#pragma unroll
  for (int r = 0; r < 8; ++r) {
    const float inv = 1.0f / l[r];
    const int t = q0 + r + hf * 8;
#pragma unroll
    for (int f = 0; f < 4; ++f) {
      const int cg = h * 64 + f * 16 + nn;
      O[((size_t)bb * Tn + t) * Cn + cg] = (bf16_t)(o[f][r] * inv);
    }
  }
}

// ---------------------------------------------------------------------------
extern "C" void kernel_launch(void* const* d_in, const int* in_sizes, int n_in,
                              void* d_out, int out_size, void* d_ws, size_t ws_size,
                              hipStream_t stream) {
  const float* x  = (const float*)d_in[0];
  const float* Wq = (const float*)d_in[1];
  const float* bq = (const float*)d_in[2];
  const float* Wk = (const float*)d_in[3];
  const float* bk = (const float*)d_in[4];
  const float* Wv = (const float*)d_in[5];
  const float* bv = (const float*)d_in[6];
  const float* Wo = (const float*)d_in[7];
  const float* bo = (const float*)d_in[8];
  float* out = (float*)d_out;

  char* ws = (char*)d_ws;
  size_t off = 0;
  auto alloc = [&](size_t bytes) -> void* {
    void* p = ws + off;
    off += (bytes + 255) & ~(size_t)255;
    return p;
  };
  bf16_t* xb   = (bf16_t*)alloc((size_t)Mn * Cn * 2);
  bf16_t* wqt  = (bf16_t*)alloc((size_t)Cn * Cn * 2);
  bf16_t* wkt  = (bf16_t*)alloc((size_t)Cn * Cn * 2);
  bf16_t* wvt  = (bf16_t*)alloc((size_t)Cn * Cn * 2);
  bf16_t* wot  = (bf16_t*)alloc((size_t)Cn * Cn * 2);
  bf16_t* qbuf = (bf16_t*)alloc((size_t)Mn * Cn * 2);
  bf16_t* kbuf = (bf16_t*)alloc((size_t)Mn * Cn * 2);
  bf16_t* vtb  = (bf16_t*)alloc((size_t)Mn * Cn * 2);
  bf16_t* attn = (bf16_t*)alloc((size_t)Mn * Cn * 2);
  (void)ws_size; (void)in_sizes; (void)n_in; (void)out_size;

  // Prep: x -> bf16, weights -> transposed bf16.
  cvt_f32_bf16<<<(Mn * Cn + 255) / 256, 256, 0, stream>>>(x, xb, Mn * Cn);
  const int wblk = (Cn * Cn + 255) / 256;
  transpose_w_bf16<<<wblk, 256, 0, stream>>>(Wq, wqt);
  transpose_w_bf16<<<wblk, 256, 0, stream>>>(Wk, wkt);
  transpose_w_bf16<<<wblk, 256, 0, stream>>>(Wv, wvt);
  transpose_w_bf16<<<wblk, 256, 0, stream>>>(Wo, wot);

  // QKV projections (each wave: 64x64 output tile).
  const int gblk = ((Mn / 64) * (Cn / 64)) / 8;   // 192
  gemm_bf16_wmma<<<gblk, 256, 0, stream>>>(xb, wqt, bq, qbuf, nullptr, MODE_QK);
  gemm_bf16_wmma<<<gblk, 256, 0, stream>>>(xb, wkt, bk, kbuf, nullptr, MODE_QK);
  gemm_bf16_wmma<<<gblk, 256, 0, stream>>>(xb, wvt, bv, vtb,  nullptr, MODE_VT);

  // Flash attention: 8 waves per block share one (b,h)'s K/V tiles via LDS.
  const int ablk = BH * (Tn / 128);               // 768
  flash_attn_wmma<<<ablk, 256, 0, stream>>>(qbuf, kbuf, vtb, attn);

  // Output projection -> fp32.
  gemm_bf16_wmma<<<gblk, 256, 0, stream>>>(attn, wot, bo, nullptr, out, MODE_F32);
}
